// DualViewCLModelPaper_88708254531756
// MI455X (gfx1250) — compile-verified, matrix operand order checked by reading, not run on hardware
//
#include <hip/hip_runtime.h>
#include <math.h>

typedef __bf16  v16bf __attribute__((ext_vector_type(16)));
typedef float   v8f   __attribute__((ext_vector_type(8)));

#define GF_RELU   1
#define GF_ACC    2
#define GF_BTRANS 4

#define CNM   1024
#define CND   1024
#define CH    256
#define CDSEM 768
#define ZSTR  260   // padded LDS row stride (floats): 4-bank shift per row

// |x| via explicit VALU bit-clear. The inline asm keeps the SLP vectorizer
// from packing the fabs chain into v2f32 ops that crash gfx1250 ISel.
__device__ __forceinline__ float fabs_valu(float x) {
  float r;
  asm("v_and_b32 %0, 0x7fffffff, %1" : "=v"(r) : "v"(x));
  return r;
}

// ---------------------------------------------------------------------------
// Generic bf16 WMMA GEMM: C(MxN) = op(A(MxK) @ B(KxN)) [+bias][+acc][relu]
// One 16x16 output tile per wave, 4 waves per block (tiles along N).
// ---------------------------------------------------------------------------
__global__ __launch_bounds__(128) void gemm_bf16_kernel(
    const float* __restrict__ A, int lda,
    const float* __restrict__ B, int ldb,
    float* __restrict__ C, int ldc,
    int M, int N, int K,
    const float* __restrict__ bias, int flags) {
  int wave = threadIdx.x >> 5;
  int lane = threadIdx.x & 31;
  int m0 = blockIdx.x * 16;
  int n0 = blockIdx.y * 64 + wave * 16;
  if (m0 >= M || n0 >= N) return;

  int row   = lane & 15;       // M index for A frag, N index for B/C frags
  int half  = lane >> 4;       // K-half selector
  int mbase = half * 8;        // C rows

  v8f c = {};
  for (int k0 = 0; k0 < K; k0 += 32) {
    v16bf a, b;
    const float* ap = A + (long)(m0 + row) * lda + k0;
#pragma unroll
    for (int i = 0; i < 8; ++i) {
      a[i]     = (__bf16)ap[half * 8 + i];
      a[8 + i] = (__bf16)ap[16 + half * 8 + i];
    }
    if (flags & GF_BTRANS) {
      // B stored as N x K row-major (logical B[k,n] = Bmem[n*ldb + k])
      const float* bp = B + (long)(n0 + row) * ldb + k0;
#pragma unroll
      for (int i = 0; i < 8; ++i) {
        b[i]     = (__bf16)bp[half * 8 + i];
        b[8 + i] = (__bf16)bp[16 + half * 8 + i];
      }
    } else {
#pragma unroll
      for (int i = 0; i < 8; ++i) {
        b[i]     = (__bf16)B[(long)(k0 + half * 8 + i) * ldb + n0 + row];
        b[8 + i] = (__bf16)B[(long)(k0 + 16 + half * 8 + i) * ldb + n0 + row];
      }
    }
    c = __builtin_amdgcn_wmma_f32_16x16x32_bf16(false, a, false, b, (short)0, c,
                                                false, false);
  }
#pragma unroll
  for (int r = 0; r < 8; ++r) {
    float v = c[r];
    if (bias) v += bias[n0 + row];
    long idx = (long)(m0 + mbase + r) * ldc + n0 + row;
    if (flags & GF_ACC)  v += C[idx];
    if (flags & GF_RELU) v = fmaxf(v, 0.f);
    C[idx] = v;
  }
}

// ---------------------------------------------------------------------------
// Elementwise / graph-scatter helpers
// ---------------------------------------------------------------------------
__global__ void fill_kernel(float* __restrict__ p, float v, int n) {
  int t = blockIdx.x * blockDim.x + threadIdx.x;
  if (t < n) p[t] = v;
}

__global__ void relu_kernel(float* __restrict__ p, int n) {
  int t = blockIdx.x * blockDim.x + threadIdx.x;
  if (t < n) p[t] = fmaxf(p[t], 0.f);
}

__global__ void count_kernel(const int* __restrict__ idx, float* __restrict__ deg, int E) {
  int t = blockIdx.x * blockDim.x + threadIdx.x;
  if (t < E) atomicAdd(&deg[idx[t]], 1.0f);
}

// agg[dst] += xw[src] * rsqrt(deg[src]) * rsqrt(deg[dst]); one block per edge
__global__ void scatter_gcn_kernel(const float* __restrict__ xw, float* __restrict__ agg,
                                   const int* __restrict__ src, const int* __restrict__ dst,
                                   const float* __restrict__ deg) {
  int e = blockIdx.x, f = threadIdx.x;
  int s = src[e], d = dst[e];
  float v = xw[(long)s * CH + f] * rsqrtf(deg[s]) * rsqrtf(deg[d]);
  atomicAdd(&agg[(long)d * CH + f], v);
}

__global__ void scatter_sum_kernel(const float* __restrict__ xs, float* __restrict__ agg,
                                   const int* __restrict__ src, const int* __restrict__ dst) {
  int e = blockIdx.x, f = threadIdx.x;
  atomicAdd(&agg[(long)dst[e] * CH + f], xs[(long)src[e] * CH + f]);
}

__global__ void sage_div_kernel(float* __restrict__ agg, const float* __restrict__ cnt) {
  int i = blockIdx.x, f = threadIdx.x;
  agg[(long)i * CH + f] /= fmaxf(cnt[i], 1.0f);
}

// out[i,f] (+)= agg[i,f] + xw[i,f]/deg[i] + b[f]
__global__ void gcn_finish_kernel(float* __restrict__ out, int ldc,
                                  const float* __restrict__ agg, const float* __restrict__ xw,
                                  const float* __restrict__ deg, const float* __restrict__ b,
                                  int acc) {
  int i = blockIdx.x, f = threadIdx.x;
  float v = agg[(long)i * CH + f] + xw[(long)i * CH + f] / deg[i] + b[f];
  long o = (long)i * ldc + f;
  if (acc) out[o] += v; else out[o] = v;
}

// ---------------------------------------------------------------------------
// Pre-pack a 256x256 weight into bf16 WMMA B-fragment order:
//   out[(((nt*8 + kk)*32 + lane)*16 + e]  (e = frag element)
//   k = kk*32 + (e>>3)*16 + (lane>>4)*8 + (e&7),  n = nt*16 + (lane&15)
// Decoder then loads one contiguous 32B v16bf per lane per fragment.
// ---------------------------------------------------------------------------
__global__ __launch_bounds__(256) void pack_w1_kernel(const float* __restrict__ w,
                                                      __bf16* __restrict__ out) {
  int t = blockIdx.x * 256 + threadIdx.x;   // 0 .. 65535
  int e    = t & 15;
  int lane = (t >> 4) & 31;
  int kk   = (t >> 9) & 7;
  int nt   = t >> 12;
  int k = kk * 32 + (e >> 3) * 16 + (lane >> 4) * 8 + (e & 7);
  int n = nt * 16 + (lane & 15);
  out[t] = (__bf16)w[(long)k * CH + n];
}

// ---------------------------------------------------------------------------
// Pairwise MLP decoder: out[m,d] += res_w * ( relu( hm[m] + hd[d]
//      + (zm[m]*zd[d])@w1p + |zm[m]-zd[d]|@w1a + b1 ) @ w2 + b2 )
// Block = 8 m-rows x 16 d-rows; 8 waves, wave w owns m-row (m0+w).
// Two independent WMMA accumulator chains (product term / abs term) to avoid
// back-to-back dependent WMMAs; C-chain is initialized with broadcast terms.
// ---------------------------------------------------------------------------
__global__ __launch_bounds__(256) void decoder_kernel(
    const float* __restrict__ zm, const float* __restrict__ zd,
    const float* __restrict__ hm, const float* __restrict__ hd,
    const __bf16* __restrict__ packP, const __bf16* __restrict__ packA,
    const float* __restrict__ b1, const float* __restrict__ w2,
    const float* __restrict__ b2, const float* __restrict__ resw,
    float* __restrict__ out) {
  __shared__ float zms[8 * ZSTR], zds[16 * ZSTR], hms[8 * ZSTR], hds[16 * ZSTR];
  __shared__ float b1s[CH], w2s[CH];
  int m0 = blockIdx.x * 8, d0 = blockIdx.y * 16;
  int tid = threadIdx.x;
  for (int i = tid; i < 8 * CH; i += 256) {
    int r = i >> 8, f = i & 255;
    zms[r * ZSTR + f] = zm[(long)(m0 + r) * CH + f];
    hms[r * ZSTR + f] = hm[(long)(m0 + r) * CH + f];
  }
  for (int i = tid; i < 16 * CH; i += 256) {
    int r = i >> 8, f = i & 255;
    zds[r * ZSTR + f] = zd[(long)(d0 + r) * CH + f];
    hds[r * ZSTR + f] = hd[(long)(d0 + r) * CH + f];
  }
  b1s[tid] = b1[tid];
  w2s[tid] = w2[tid];
  __syncthreads();

  int wave = tid >> 5, lane = tid & 31;
  int row   = lane & 15;    // A frag: pair row (dj). B/C frag: hidden col
  int half  = lane >> 4;
  int mbase = half * 8;     // C frag rows (dj = mbase + r)
  const float* zmr = &zms[wave * ZSTR];
  const float* zdr = &zds[row * ZSTR];

  float part[8];
#pragma unroll
  for (int r = 0; r < 8; ++r) part[r] = 0.f;

  for (int nt = 0; nt < 16; ++nt) {
    int n0 = nt * 16;
    int n  = n0 + row;
    v8f c;   // product-term chain, carries the broadcast init
    v8f c2 = {};  // abs-term chain (independent)
#pragma unroll
    for (int r = 0; r < 8; ++r)
      c[r] = hms[wave * ZSTR + n] + hds[(mbase + r) * ZSTR + n] + b1s[n];

    for (int kk8 = 0; kk8 < 8; ++kk8) {
      int kk = kk8 * 32;
      // coalesced 32B fragment loads from the pre-packed bf16 weights
      v16bf bp = *(const v16bf*)(packP + (((long)nt * 8 + kk8) * 32 + lane) * 16);
      v16bf ba = *(const v16bf*)(packA + (((long)nt * 8 + kk8) * 32 + lane) * 16);
      v16bf am, aa;
#pragma unroll
      for (int i = 0; i < 8; ++i) {
        int k1 = kk + half * 8 + i;
        int k2 = kk + 16 + half * 8 + i;
        float zm1 = zmr[k1], zd1 = zdr[k1];
        float zm2 = zmr[k2], zd2 = zdr[k2];
        am[i]     = (__bf16)(zm1 * zd1);
        am[8 + i] = (__bf16)(zm2 * zd2);
        aa[i]     = (__bf16)fabs_valu(zm1 - zd1);
        aa[8 + i] = (__bf16)fabs_valu(zm2 - zd2);
      }
      c  = __builtin_amdgcn_wmma_f32_16x16x32_bf16(false, am, false, bp, (short)0, c,
                                                   false, false);
      c2 = __builtin_amdgcn_wmma_f32_16x16x32_bf16(false, aa, false, ba, (short)0, c2,
                                                   false, false);
    }
    float w2v = w2s[n];
#pragma unroll
    for (int r = 0; r < 8; ++r) part[r] += fmaxf(c[r] + c2[r], 0.f) * w2v;
  }

  // reduce the 16 hidden-column lanes within each half-wave
#pragma unroll
  for (int r = 0; r < 8; ++r) {
    float v = part[r];
    v += __shfl_xor(v, 1, 32);
    v += __shfl_xor(v, 2, 32);
    v += __shfl_xor(v, 4, 32);
    v += __shfl_xor(v, 8, 32);
    part[r] = v;
  }
  float rw = resw[0], bb2 = b2[0];
  if (row == 0) {
#pragma unroll
    for (int r = 0; r < 8; ++r) {
      int dj = mbase + r;
      out[(long)(m0 + wave) * CND + d0 + dj] += rw * (part[r] + bb2);
    }
  }
}

// ---------------------------------------------------------------------------
// Host orchestration
// ---------------------------------------------------------------------------
extern "C" void kernel_launch(void* const* d_in, const int* in_sizes, int n_in,
                              void* d_out, int out_size, void* d_ws, size_t ws_size,
                              hipStream_t stream) {
  (void)n_in; (void)out_size; (void)ws_size;
  const int NM = CNM, ND = CND, H = CH, DSEM = CDSEM;
  const int E = in_sizes[3] / 2;

  const float* x_m_sim = (const float*)d_in[0];
  const float* x_m_sem = (const float*)d_in[1];
  const float* x_d_sim = (const float*)d_in[2];
  const int*   e_mm    = (const int*)d_in[3];
  const int*   e_dd    = (const int*)d_in[4];
  const int*   e_md    = (const int*)d_in[5];
  auto P = [&](int i) { return (const float*)d_in[6 + i]; };
  // param indices (setup_inputs insertion order)
  const float *pm_sim_w1=P(0),*pm_sim_b1=P(1),*pm_sim_w2=P(2),*pm_sim_b2=P(3);
  const float *pm_sem_w1=P(4),*pm_sem_b1=P(5),*pm_sem_w2=P(6),*pm_sem_b2=P(7);
  const float *pd_sim_w1=P(8),*pd_sim_b1=P(9),*pd_sim_w2=P(10),*pd_sim_b2=P(11);
  const float *g1_md_wl=P(12),*g1_md_bl=P(13),*g1_md_wr=P(14);
  const float *g1_dm_wl=P(15),*g1_dm_bl=P(16),*g1_dm_wr=P(17);
  const float *g1_mm_w=P(18),*g1_mm_b=P(19),*g1_mmr_w=P(20),*g1_mmr_b=P(21);
  const float *g1_dd_w=P(22),*g1_dd_b=P(23),*g1_ddr_w=P(24),*g1_ddr_b=P(25);
  const float *g2_md_wl=P(26),*g2_md_bl=P(27),*g2_md_wr=P(28);
  const float *g2_dm_wl=P(29),*g2_dm_bl=P(30),*g2_dm_wr=P(31);
  const float *g2_mm_w=P(32),*g2_mm_b=P(33),*g2_mmr_w=P(34),*g2_mmr_b=P(35);
  const float *g2_dd_w=P(36),*g2_dd_b=P(37),*g2_ddr_w=P(38),*g2_ddr_b=P(39);
  const float *fm_w=P(40),*fm_b=P(41),*fd_w=P(42),*fd_b=P(43);
  const float *W_bi=P(44);
  const float *dec_w1=P(45),*dec_b1=P(46),*dec_w2=P(47),*dec_b2=P(48),*res_w=P(49);

  float* wsp = (float*)d_ws;
  auto alloc = [&](size_t n) { float* p = wsp; wsp += n; return p; };
  float* m_sim  = alloc((size_t)NM * H);
  float* m_sem  = alloc((size_t)NM * H);
  float* d_feat = alloc((size_t)ND * H);
  float* hbuf   = alloc((size_t)NM * H);
  float* xw     = alloc((size_t)NM * H);
  float* agg    = alloc((size_t)NM * H);
  float* m1     = alloc((size_t)NM * H);
  float* d1     = alloc((size_t)ND * H);
  float* zcat_m = alloc((size_t)NM * 2 * H);
  float* zcat_d = alloc((size_t)ND * 2 * H);
  float* zm     = alloc((size_t)NM * H);
  float* zd     = alloc((size_t)ND * H);
  float* hm     = alloc((size_t)NM * H);
  float* hd     = alloc((size_t)ND * H);
  float* t1     = alloc((size_t)NM * H);
  float* deg_mm_f = alloc(NM); float* deg_mm_r = alloc(NM);
  float* deg_dd_f = alloc(ND); float* deg_dd_r = alloc(ND);
  float* cnt_md_d = alloc(ND); float* cnt_md_m = alloc(NM);
  __bf16* packP = (__bf16*)alloc(32768);  // 65536 bf16
  __bf16* packA = (__bf16*)alloc(32768);  // 65536 bf16

  auto gemm = [&](const float* A, int lda, const float* B, int ldb, float* C, int ldc,
                  int M, int N, int K, const float* bias, int flags) {
    dim3 g(M / 16, N / 64);
    gemm_bf16_kernel<<<g, 128, 0, stream>>>(A, lda, B, ldb, C, ldc, M, N, K, bias, flags);
  };
  auto fill = [&](float* p, float v, int n) {
    fill_kernel<<<(n + 255) / 256, 256, 0, stream>>>(p, v, n);
  };

  // ---- degrees / counts (depend only on edge lists) ----
  fill(deg_mm_f, 1.f, NM); count_kernel<<<(E + 255) / 256, 256, 0, stream>>>(e_mm + E, deg_mm_f, E);
  fill(deg_mm_r, 1.f, NM); count_kernel<<<(E + 255) / 256, 256, 0, stream>>>(e_mm,     deg_mm_r, E);
  fill(deg_dd_f, 1.f, ND); count_kernel<<<(E + 255) / 256, 256, 0, stream>>>(e_dd + E, deg_dd_f, E);
  fill(deg_dd_r, 1.f, ND); count_kernel<<<(E + 255) / 256, 256, 0, stream>>>(e_dd,     deg_dd_r, E);
  fill(cnt_md_d, 0.f, ND); count_kernel<<<(E + 255) / 256, 256, 0, stream>>>(e_md + E, cnt_md_d, E);
  fill(cnt_md_m, 0.f, NM); count_kernel<<<(E + 255) / 256, 256, 0, stream>>>(e_md,     cnt_md_m, E);

  // ---- pre-pack decoder pairwise weights into bf16 fragment order ----
  pack_w1_kernel<<<256, 256, 0, stream>>>(dec_w1 + (size_t)2 * H * H, packP);  // w1p
  pack_w1_kernel<<<256, 256, 0, stream>>>(dec_w1 + (size_t)3 * H * H, packA);  // w1a

  // ---- input projections (MLP2) ----
  gemm(x_m_sim, NM,   pm_sim_w1, H, hbuf, H, NM, H, NM,   pm_sim_b1, GF_RELU);
  gemm(hbuf,    H,    pm_sim_w2, H, m_sim, H, NM, H, H,   pm_sim_b2, 0);
  gemm(x_m_sem, DSEM, pm_sem_w1, H, hbuf, H, NM, H, DSEM, pm_sem_b1, GF_RELU);
  gemm(hbuf,    H,    pm_sem_w2, H, m_sem, H, NM, H, H,   pm_sem_b2, 0);
  gemm(x_d_sim, ND,   pd_sim_w1, H, hbuf, H, ND, H, ND,   pd_sim_b1, GF_RELU);
  gemm(hbuf,    H,    pd_sim_w2, H, d_feat, H, ND, H, H,  pd_sim_b2, 0);

  auto run_gcn = [&](const float* x, const float* wgt, const float* bias,
                     const int* src, const int* dst, const float* deg, int n,
                     float* out, int ldc, int acc) {
    gemm(x, H, wgt, H, xw, H, n, H, H, nullptr, 0);
    fill(agg, 0.f, n * H);
    scatter_gcn_kernel<<<E, H, 0, stream>>>(xw, agg, src, dst, deg);
    gcn_finish_kernel<<<n, H, 0, stream>>>(out, ldc, agg, xw, deg, bias, acc);
  };
  auto run_sage = [&](const float* xs, const float* xd, const float* wl, const float* bl,
                      const float* wr, const int* src, const int* dst, const float* cnt,
                      int ndst, float* out, int ldc) {
    fill(agg, 0.f, ndst * H);
    scatter_sum_kernel<<<E, H, 0, stream>>>(xs, agg, src, dst);
    sage_div_kernel<<<ndst, H, 0, stream>>>(agg, cnt);
    gemm(agg, H, wl, H, out, ldc, ndst, H, H, bl, GF_ACC);
    gemm(xd,  H, wr, H, out, ldc, ndst, H, H, nullptr, GF_ACC);
  };

  // ---- branch "sim" (no md edges) ----
  run_gcn(m_sim, g1_mm_w, g1_mm_b,  e_mm,     e_mm + E, deg_mm_f, NM, m1, H, 0);
  run_gcn(m_sim, g1_mmr_w, g1_mmr_b, e_mm + E, e_mm,    deg_mm_r, NM, m1, H, 1);
  run_gcn(d_feat, g1_dd_w, g1_dd_b,  e_dd,     e_dd + E, deg_dd_f, ND, d1, H, 0);
  run_gcn(d_feat, g1_ddr_w, g1_ddr_b, e_dd + E, e_dd,    deg_dd_r, ND, d1, H, 1);
  relu_kernel<<<(NM * H + 255) / 256, 256, 0, stream>>>(m1, NM * H);
  relu_kernel<<<(ND * H + 255) / 256, 256, 0, stream>>>(d1, ND * H);
  run_gcn(m1, g2_mm_w, g2_mm_b,  e_mm,     e_mm + E, deg_mm_f, NM, zcat_m, 2 * H, 0);
  run_gcn(m1, g2_mmr_w, g2_mmr_b, e_mm + E, e_mm,    deg_mm_r, NM, zcat_m, 2 * H, 1);
  run_gcn(d1, g2_dd_w, g2_dd_b,  e_dd,     e_dd + E, deg_dd_f, ND, zcat_d, 2 * H, 0);
  run_gcn(d1, g2_ddr_w, g2_ddr_b, e_dd + E, e_dd,    deg_dd_r, ND, zcat_d, 2 * H, 1);

  // ---- branch "sem" (with md edges) ----
  run_gcn(m_sem, g1_mm_w, g1_mm_b,  e_mm,     e_mm + E, deg_mm_f, NM, m1, H, 0);
  run_gcn(m_sem, g1_mmr_w, g1_mmr_b, e_mm + E, e_mm,    deg_mm_r, NM, m1, H, 1);
  run_sage(d_feat, m_sem, g1_dm_wl, g1_dm_bl, g1_dm_wr, e_md + E, e_md, cnt_md_m, NM, m1, H);
  run_gcn(d_feat, g1_dd_w, g1_dd_b,  e_dd,     e_dd + E, deg_dd_f, ND, d1, H, 0);
  run_gcn(d_feat, g1_ddr_w, g1_ddr_b, e_dd + E, e_dd,    deg_dd_r, ND, d1, H, 1);
  run_sage(m_sem, d_feat, g1_md_wl, g1_md_bl, g1_md_wr, e_md, e_md + E, cnt_md_d, ND, d1, H);
  relu_kernel<<<(NM * H + 255) / 256, 256, 0, stream>>>(m1, NM * H);
  relu_kernel<<<(ND * H + 255) / 256, 256, 0, stream>>>(d1, ND * H);
  run_gcn(m1, g2_mm_w, g2_mm_b,  e_mm,     e_mm + E, deg_mm_f, NM, zcat_m + H, 2 * H, 0);
  run_gcn(m1, g2_mmr_w, g2_mmr_b, e_mm + E, e_mm,    deg_mm_r, NM, zcat_m + H, 2 * H, 1);
  run_sage(d1, m1, g2_dm_wl, g2_dm_bl, g2_dm_wr, e_md + E, e_md, cnt_md_m, NM, zcat_m + H, 2 * H);
  run_gcn(d1, g2_dd_w, g2_dd_b,  e_dd,     e_dd + E, deg_dd_f, ND, zcat_d + H, 2 * H, 0);
  run_gcn(d1, g2_ddr_w, g2_ddr_b, e_dd + E, e_dd,    deg_dd_r, ND, zcat_d + H, 2 * H, 1);
  run_sage(m1, d1, g2_md_wl, g2_md_bl, g2_md_wr, e_md, e_md + E, cnt_md_d, ND, zcat_d + H, 2 * H);

  // ---- fusion ----
  gemm(zcat_m, 2 * H, fm_w, H, zm, H, NM, H, 2 * H, fm_b, GF_RELU);
  gemm(zcat_d, 2 * H, fd_w, H, zd, H, ND, H, 2 * H, fd_b, GF_RELU);

  // ---- bilinear logits -> d_out ----
  gemm(zm, H, W_bi, H, t1, H, NM, H, H, nullptr, 0);
  gemm(t1, H, zd, H, (float*)d_out, ND, NM, ND, H, nullptr, GF_BTRANS);

  // ---- pairwise MLP decoder (residual add into d_out) ----
  gemm(zm, H, dec_w1,                 H, hm, H, NM, H, H, nullptr, 0);  // w1m
  gemm(zd, H, dec_w1 + (size_t)H * H, H, hd, H, ND, H, H, nullptr, 0);  // w1d
  decoder_kernel<<<dim3(NM / 8, ND / 16), 256, 0, stream>>>(
      zm, zd, hm, hd, packP, packA, dec_b1, dec_w2, dec_b2, res_w, (float*)d_out);
}
